// MoEDecoderLayer_35107062678430
// MI455X (gfx1250) — matrix operand order repaired and checked by reading, not compiled, not run on hardware
//
#include <hip/hip_runtime.h>
#include <hip/hip_bf16.h>

// ---------------- problem constants (match reference) ----------------
#define Hh   1024
#define Ii   512
#define Ee   64
#define CAP  1024
#define EPSV 1e-06f

typedef __attribute__((ext_vector_type(16))) __bf16 v16bf;
typedef __attribute__((ext_vector_type(8)))  float  v8f;
typedef __attribute__((ext_vector_type(4)))  unsigned int u32x4;
typedef __attribute__((ext_vector_type(8)))  int i32x8;
typedef __attribute__((ext_vector_type(4)))  int i32x4;

// ---------------- bf16 conversion: single packed convert via inline asm ----------------
// (clang-22 has no __builtin_amdgcn_cvt_pk_bf16_f32 for gfx1250, but the
//  assembler accepts the instruction — the compiler itself emits it for
//  scalar __bf16 truncation with a zero hi operand.)
__device__ __forceinline__ unsigned int pack2bf(float lo, float hi) {
  unsigned r;
  asm("v_cvt_pk_bf16_f32 %0, %1, %2" : "=v"(r) : "v"(lo), "v"(hi));
  return r;
}
__device__ __forceinline__ unsigned short f2bf(float f) {
  return (unsigned short)pack2bf(f, 0.f);
}
__device__ __forceinline__ float bf2f(unsigned short b) {
  return __uint_as_float(((unsigned)b) << 16);
}
__device__ __forceinline__ float fast_sig(float x) {          // 1/(1+exp(-x)) via v_rcp
  return __builtin_amdgcn_rcpf(1.f + __expf(-x));
}
__device__ __forceinline__ float silu(float g) { return g * fast_sig(g); }

// ---------------- Tensor Data Mover (TDM) helpers ----------------
// D# per CDNA5 ISA: group0 = {flags/count, lds_addr,
// global_addr lo, global_addr hi | type=2}; group1 packs data_size=2B,
// pad_enable, pad_interval=7 (1 DWORD pad per 256 DWORDs stored -> LDS row
// strides 257/514 u32 are bank-conflict free), tensor/tile dims, stride.
#define HAVE_TDM __has_builtin(__builtin_amdgcn_tensor_load_to_lds)

__device__ __forceinline__ void tdm_wait_all() {
#if __has_builtin(__builtin_amdgcn_s_wait_tensorcnt)
  __builtin_amdgcn_s_wait_tensorcnt((short)0);
#else
  asm volatile("s_wait_tensorcnt 0x0" ::: "memory");
#endif
}

__device__ __forceinline__ u32x4 tdm_g0(unsigned flags, unsigned ldsOff, const void* ga) {
  unsigned long long a = (unsigned long long)ga;
  u32x4 g;
  g[0] = 1u | flags;                                   // count=1 (+gather flags)
  g[1] = ldsOff;                                       // LDS byte address
  g[2] = (unsigned)a;                                  // global_addr[31:0]
  g[3] = (unsigned)((a >> 32) & 0x1FFFFFFull) | (2u << 30);  // addr[56:32] | type=2
  return g;
}
__device__ __forceinline__ i32x8 tdm_g1(unsigned td0, unsigned td1, unsigned tl0,
                                        unsigned tl1, unsigned long long s0) {
  i32x8 g;
  g[0] = (int)((1u << 16) | (1u << 20) | (7u << 22));  // data_size=2B | pad_enable | pad_interval=256DW
  g[1] = (int)((td0 & 0xFFFFu) << 16);                 // tensor_dim0[15:0]
  g[2] = (int)((td0 >> 16) | ((td1 & 0xFFFFu) << 16)); // tensor_dim0[31:16] | tensor_dim1[15:0]
  g[3] = (int)((td1 >> 16) | ((tl0 & 0xFFFFu) << 16)); // tensor_dim1[31:16] | tile_dim0
  g[4] = (int)(tl1 & 0xFFFFu);                         // tile_dim1 (tile_dim2=0)
  g[5] = (int)(s0 & 0xFFFFFFFFull);                    // tensor_dim0_stride[31:0]
  g[6] = (int)((s0 >> 32) & 0xFFFFull);                // stride[47:32] (dim1_stride=0)
  g[7] = 0;
  return g;
}
__device__ __forceinline__ void tdm_issue(u32x4 g0, i32x8 g1, i32x4 g2, i32x4 g3) {
#if HAVE_TDM
#if __clang_major__ >= 23
  i32x8 z8 = {0, 0, 0, 0, 0, 0, 0, 0};
  __builtin_amdgcn_tensor_load_to_lds(g0, g1, g2, g3, z8, 0);
#else
  __builtin_amdgcn_tensor_load_to_lds(g0, g1, g2, g3, 0);
#endif
#endif
}

// contiguous 2D tile (rowElems bf16 per row, nrows rows, row stride in elems)
__device__ __forceinline__ void tdm_load_2d(void* lds, const unsigned short* ga,
                                            unsigned rowElems, unsigned nrows,
                                            unsigned strideElems, int lane) {
#if HAVE_TDM
  (void)lane;
  i32x4 z = {0, 0, 0, 0};
  tdm_issue(tdm_g0(0u, (unsigned)(unsigned long long)lds, ga),
            tdm_g1(rowElems, nrows, rowElems, nrows, strideElems), z, z);
#else
  unsigned* L = (unsigned*)lds;
  const unsigned* G = (const unsigned*)(const void*)ga;
  unsigned ppr = rowElems >> 1, st = ppr + (ppr >> 8);
  for (unsigned i = lane; i < nrows * ppr; i += 32) {
    unsigned r = i / ppr, p = i - r * ppr;
    L[r * st + p + (p >> 8)] = G[(size_t)r * (strideElems >> 1) + p];
  }
#endif
}

// gather-mode tile: up to 16 rows selected by 16-bit indices (MoE token gather)
__device__ __forceinline__ void tdm_gather16(void* lds, const unsigned short* ga,
                                             unsigned rowElems, unsigned nIdx,
                                             unsigned totRows, unsigned strideElems,
                                             const unsigned* pidx, int lane) {
#if HAVE_TDM
  (void)lane;
  i32x4 g2 = {(int)pidx[0], (int)pidx[1], (int)pidx[2], (int)pidx[3]};
  i32x4 g3 = {(int)pidx[4], (int)pidx[5], (int)pidx[6], (int)pidx[7]};
  tdm_issue(tdm_g0(1u << 31, (unsigned)(unsigned long long)lds, ga),  // gather_mode, 16b idx
            tdm_g1(rowElems, totRows, rowElems, nIdx, strideElems), g2, g3);
#else
  unsigned* L = (unsigned*)lds;
  const unsigned* G = (const unsigned*)(const void*)ga;
  unsigned ppr = rowElems >> 1, st = ppr + (ppr >> 8);
  for (unsigned i = lane; i < nIdx * ppr; i += 32) {
    unsigned r = i / ppr, p = i - r * ppr;
    unsigned tok = (pidx[r >> 1] >> ((r & 1) * 16)) & 0xFFFFu;
    L[r * st + p + (p >> 8)] = G[(size_t)tok * (strideElems >> 1) + p];
  }
#endif
}

// ---------------- WMMA fragment builders ----------------
// A (16x32 bf16) from a padded LDS panel: row stride strideU32, pair index p
// gets +1 skew per 256 u32 (TDM pad). Lane L: m=L&15, grp=L>>4.
__device__ __forceinline__ void load_afrag_pan(const unsigned* pan, int strideU32,
                                               int lane, int basePair, unsigned* au) {
  int m = lane & 15, grp = lane >> 4;
  const unsigned* row = pan + m * strideU32;
#pragma unroll
  for (int j = 0; j < 4; ++j) { int p = basePair + grp * 4 + j;     au[j]     = row[p + (p >> 8)]; }
#pragma unroll
  for (int j = 0; j < 4; ++j) { int p = basePair + 8 + grp * 4 + j; au[4 + j] = row[p + (p >> 8)]; }
}
// B (32x16 bf16) from LDS stored column-major [cols][17 u32]
__device__ __forceinline__ void load_bfrag(const unsigned* lB, int ni, int lane, unsigned* bu) {
  const unsigned* col = lB + (ni * 16 + (lane & 15)) * 17;
  int grp = lane >> 4;
#pragma unroll
  for (int j = 0; j < 8; ++j) bu[j] = col[(j >> 2) * 8 + grp * 4 + (j & 3)];
}
__device__ __forceinline__ v8f wmma_bf16(const unsigned* au, const unsigned* bu, v8f c) {
  union { unsigned u[8]; v16bf v; } A, B;
#pragma unroll
  for (int j = 0; j < 8; ++j) { A.u[j] = au[j]; B.u[j] = bu[j]; }
  return __builtin_amdgcn_wmma_f32_16x16x32_bf16(false, A.v, false, B.v, (short)0, c, false, false);
}

// ---------------- 1) RMSNorm -> packed bf16 x ----------------
__global__ void k_rmsnorm(const float* __restrict__ hs, const float* __restrict__ w,
                          unsigned* __restrict__ xb16) {
  int t = blockIdx.x;
  const float* row = hs + (size_t)t * Hh;
  __shared__ float red[8];
  float s = 0.f;
  for (int h = threadIdx.x; h < Hh; h += 256) { float v = row[h]; s += v * v; }
#pragma unroll
  for (int off = 16; off; off >>= 1) s += __shfl_xor(s, off, 32);
  if ((threadIdx.x & 31) == 0) red[threadIdx.x >> 5] = s;
  __syncthreads();
  float tot = 0.f;
#pragma unroll
  for (int i = 0; i < 8; ++i) tot += red[i];
  float rs = rsqrtf(tot / (float)Hh + EPSV);
  unsigned* dst = xb16 + (size_t)t * (Hh / 2);
  for (int p = threadIdx.x; p < Hh / 2; p += 256) {
    int h = 2 * p;
    dst[p] = pack2bf(row[h] * rs * w[h], row[h + 1] * rs * w[h + 1]);
  }
}

__global__ void k_zero_counts(int* __restrict__ c) {
  if (threadIdx.x < Ee) c[threadIdx.x] = 0;
}

// ---------------- 2) Gating: logits, top-2 softmax, capacity dispatch ----------------
__global__ void k_gate(const unsigned* __restrict__ xb16, const float* __restrict__ gw,
                       int* __restrict__ counts, int* __restrict__ etok,
                       float* __restrict__ ewt, int T) {
  int wv = threadIdx.x >> 5, lane = threadIdx.x & 31;
  int t = blockIdx.x * 8 + wv;
  if (t >= T) return;
  const unsigned* xr = xb16 + (size_t)t * (Hh / 2);
  float xv[32];
#pragma unroll
  for (int i = 0; i < 16; ++i) {
    unsigned u = xr[lane + 32 * i];
    xv[2 * i]     = bf2f((unsigned short)(u & 0xFFFFu));
    xv[2 * i + 1] = bf2f((unsigned short)(u >> 16));
  }
  float m0 = 0.f, m1 = 0.f;
  for (int e = 0; e < Ee; ++e) {
    const float2* g2 = (const float2*)(gw + (size_t)e * Hh);
    float s = 0.f;
#pragma unroll 4
    for (int i = 0; i < 16; ++i) {
      float2 gg = g2[lane + 32 * i];
      s += xv[2 * i] * gg.x + xv[2 * i + 1] * gg.y;
    }
#pragma unroll
    for (int off = 16; off; off >>= 1) s += __shfl_xor(s, off, 32);
    if (lane == (e & 31)) { if (e < 32) m0 = s; else m1 = s; }
  }
  const float NEGBIG = -3.402823466e+38f;
  float bv; int bi;
  if (m0 >= m1) { bv = m0; bi = lane; } else { bv = m1; bi = lane + 32; }
#pragma unroll
  for (int off = 16; off; off >>= 1) {
    float ov = __shfl_xor(bv, off, 32); int oi = __shfl_xor(bi, off, 32);
    if (ov > bv || (ov == bv && oi < bi)) { bv = ov; bi = oi; }
  }
  float c0 = (bi == lane)      ? NEGBIG : m0;
  float c1 = (bi == lane + 32) ? NEGBIG : m1;
  float bv2; int bi2;
  if (c0 >= c1) { bv2 = c0; bi2 = lane; } else { bv2 = c1; bi2 = lane + 32; }
#pragma unroll
  for (int off = 16; off; off >>= 1) {
    float ov = __shfl_xor(bv2, off, 32); int oi = __shfl_xor(bi2, off, 32);
    if (ov > bv2 || (ov == bv2 && oi < bi2)) { bv2 = ov; bi2 = oi; }
  }
  if (lane == 0) {
    float w1 = fast_sig(bv - bv2);                 // p1/(p1+p2)
    float w2 = 1.f - w1;
    int s1 = atomicAdd(&counts[bi], 1);
    if (s1 < CAP) { etok[bi * CAP + s1] = t; ewt[bi * CAP + s1] = w1; }
    int s2 = atomicAdd(&counts[bi2], 1);
    if (s2 < CAP) { etok[bi2 * CAP + s2] = t; ewt[bi2 * CAP + s2] = w2; }
  }
}

// ---------------- 3) Shared expert up (16x128 tile, A panel via TDM) ----------------
__global__ void __launch_bounds__(256) k_shared_up(const unsigned* __restrict__ xb16,
    const float* __restrict__ swg, const float* __restrict__ swu,
    unsigned short* __restrict__ hsh) {
  const int NT = Ii / 128;                       // 4
  int mt = blockIdx.x / NT, nt = blockIdx.x % NT;
  __shared__ unsigned pA[16 * 514];              // 16 rows x 1024 bf16 (+pad)
  __shared__ unsigned lBg[128 * 17];
  __shared__ unsigned lBu[128 * 17];
  int tid = threadIdx.x, wv = tid >> 5, lane = tid & 31;
  int ni = wv, cbase = nt * 128;
  if (wv == 0)
    tdm_load_2d(pA, (const unsigned short*)(xb16 + (size_t)(mt * 16) * (Hh / 2)),
                Hh, 16, Hh, lane);
  int c4 = tid & 31, pp = tid >> 5;              // 4 cols x pairs {pp, pp+8}
  const float* g0p = swg + (size_t)(2 * pp) * Ii + cbase + 4 * c4;
  const float* u0p = swu + (size_t)(2 * pp) * Ii + cbase + 4 * c4;
  float4 rg[2][2], ru[2][2];
#pragma unroll
  for (int h = 0; h < 2; ++h) {
    const float* gq = g0p + (size_t)(16 * h) * Ii;
    const float* uq = u0p + (size_t)(16 * h) * Ii;
    rg[h][0] = *(const float4*)gq; rg[h][1] = *(const float4*)(gq + Ii);
    ru[h][0] = *(const float4*)uq; ru[h][1] = *(const float4*)(uq + Ii);
  }
  if (wv == 0) tdm_wait_all();
  __syncthreads();
  v8f cg = {}, cu = {};
  for (int k0 = 0; k0 < Hh; k0 += 32) {
#pragma unroll
    for (int h = 0; h < 2; ++h) {
      int q = pp + 8 * h;
      unsigned* dg = &lBg[(4 * c4) * 17 + q];
      unsigned* du = &lBu[(4 * c4) * 17 + q];
      dg[0]  = pack2bf(rg[h][0].x, rg[h][1].x); dg[17] = pack2bf(rg[h][0].y, rg[h][1].y);
      dg[34] = pack2bf(rg[h][0].z, rg[h][1].z); dg[51] = pack2bf(rg[h][0].w, rg[h][1].w);
      du[0]  = pack2bf(ru[h][0].x, ru[h][1].x); du[17] = pack2bf(ru[h][0].y, ru[h][1].y);
      du[34] = pack2bf(ru[h][0].z, ru[h][1].z); du[51] = pack2bf(ru[h][0].w, ru[h][1].w);
    }
    __syncthreads();
    if (k0 + 32 < Hh) {
#pragma unroll
      for (int h = 0; h < 2; ++h) {
        const float* gq = g0p + (size_t)(k0 + 32 + 16 * h) * Ii;
        const float* uq = u0p + (size_t)(k0 + 32 + 16 * h) * Ii;
        rg[h][0] = *(const float4*)gq; rg[h][1] = *(const float4*)(gq + Ii);
        ru[h][0] = *(const float4*)uq; ru[h][1] = *(const float4*)(uq + Ii);
      }
    }
    unsigned au[8], bg[8], bu[8];
    load_afrag_pan(pA, 514, lane, k0 >> 1, au);
    load_bfrag(lBg, ni, lane, bg);
    load_bfrag(lBu, ni, lane, bu);
    cg = wmma_bf16(au, bg, cg);
    cu = wmma_bf16(au, bu, cu);
    __syncthreads();
  }
  int grp = lane >> 4, nn = lane & 15;
  int col = cbase + ni * 16 + nn;
#pragma unroll
  for (int j = 0; j < 8; ++j) {
    int r = j + grp * 8;
    hsh[(size_t)(mt * 16 + r) * Ii + col] = f2bf(silu(cg[j]) * cu[j]);
  }
}

// ---------------- 4) Shared expert down + residual (initializes d_out) ----------------
__global__ void __launch_bounds__(256) k_shared_down(const unsigned short* __restrict__ hsh,
    const float* __restrict__ swd, const float* __restrict__ resid, float* __restrict__ out) {
  const int NT = Hh / 128;                       // 8
  int mt = blockIdx.x / NT, nt = blockIdx.x % NT;
  __shared__ unsigned pA[16 * 257];              // 16 rows x 512 bf16 (+pad)
  __shared__ unsigned lB[128 * 17];
  int tid = threadIdx.x, wv = tid >> 5, lane = tid & 31;
  int ni = wv, cbase = nt * 128;
  if (wv == 0)
    tdm_load_2d(pA, hsh + (size_t)(mt * 16) * Ii, Ii, 16, Ii, lane);
  int c4 = tid & 31, pp = tid >> 5;
  const float* b0p = swd + (size_t)(2 * pp) * Hh + cbase + 4 * c4;
  float4 rb[2][2];
#pragma unroll
  for (int h = 0; h < 2; ++h) {
    const float* bq = b0p + (size_t)(16 * h) * Hh;
    rb[h][0] = *(const float4*)bq; rb[h][1] = *(const float4*)(bq + Hh);
  }
  if (wv == 0) tdm_wait_all();
  __syncthreads();
  v8f acc = {};
  for (int k0 = 0; k0 < Ii; k0 += 32) {
#pragma unroll
    for (int h = 0; h < 2; ++h) {
      int q = pp + 8 * h;
      unsigned* db = &lB[(4 * c4) * 17 + q];
      db[0]  = pack2bf(rb[h][0].x, rb[h][1].x); db[17] = pack2bf(rb[h][0].y, rb[h][1].y);
      db[34] = pack2bf(rb[h][0].z, rb[h][1].z); db[51] = pack2bf(rb[h][0].w, rb[h][1].w);
    }
    __syncthreads();
    if (k0 + 32 < Ii) {
#pragma unroll
      for (int h = 0; h < 2; ++h) {
        const float* bq = b0p + (size_t)(k0 + 32 + 16 * h) * Hh;
        rb[h][0] = *(const float4*)bq; rb[h][1] = *(const float4*)(bq + Hh);
      }
    }
    unsigned au[8], bu[8];
    load_afrag_pan(pA, 257, lane, k0 >> 1, au);
    load_bfrag(lB, ni, lane, bu);
    acc = wmma_bf16(au, bu, acc);
    __syncthreads();
  }
  int grp = lane >> 4, nn = lane & 15;
  int col = cbase + ni * 16 + nn;
#pragma unroll
  for (int j = 0; j < 8; ++j) {
    int r = j + grp * 8;
    size_t o = (size_t)(mt * 16 + r) * Hh + col;
    out[o] = resid[o] + acc[j];
  }
}

// ---------------- 5) Routed experts up (A gathered by TDM gather-mode) ----------------
__global__ void __launch_bounds__(256) k_expert_up(const unsigned* __restrict__ xb16,
    const float* __restrict__ w13, const int* __restrict__ counts,
    const int* __restrict__ etok, unsigned short* __restrict__ hact, int T) {
  const int NT = Ii / 128;                       // 4
  const int MT = CAP / 16;                       // 64
  int b = blockIdx.x;
  int e = b / (MT * NT); int rem = b % (MT * NT);
  int mt = rem / NT, nt = rem % NT;
  int n = counts[e]; if (n > CAP) n = CAP;
  if (mt * 16 >= n) return;
  int nrem = n - mt * 16; if (nrem > 16) nrem = 16;

  __shared__ unsigned pA[16 * 514];
  __shared__ unsigned lBg[128 * 17];
  __shared__ unsigned lBu[128 * 17];
  int tid = threadIdx.x, wv = tid >> 5, lane = tid & 31;
  int ni = wv, cbase = nt * 128;
  if (wv == 0) {
    const int* lst = etok + e * CAP + mt * 16;
    unsigned pidx[8];
#pragma unroll
    for (int i = 0; i < 8; ++i) {
      int a0 = (2 * i     < nrem) ? lst[2 * i]     : 0;
      int a1 = (2 * i + 1 < nrem) ? lst[2 * i + 1] : 0;
      pidx[i] = (unsigned)__builtin_amdgcn_readfirstlane((a0 & 0xFFFF) | (a1 << 16));
    }
    tdm_gather16(pA, (const unsigned short*)xb16, Hh, (unsigned)nrem,
                 (unsigned)T, Hh, pidx, lane);
  }
  int c4 = tid & 31, pp = tid >> 5;
  const float* g0p = w13 + (size_t)e * Hh * (2 * Ii) + (size_t)(2 * pp) * (2 * Ii) + cbase + 4 * c4;
  const float* u0p = g0p + Ii;
  float4 rg[2][2], ru[2][2];
#pragma unroll
  for (int h = 0; h < 2; ++h) {
    const float* gq = g0p + (size_t)(16 * h) * (2 * Ii);
    const float* uq = u0p + (size_t)(16 * h) * (2 * Ii);
    rg[h][0] = *(const float4*)gq; rg[h][1] = *(const float4*)(gq + 2 * Ii);
    ru[h][0] = *(const float4*)uq; ru[h][1] = *(const float4*)(uq + 2 * Ii);
  }
  if (wv == 0) tdm_wait_all();
  __syncthreads();
  v8f cg = {}, cu = {};
  for (int k0 = 0; k0 < Hh; k0 += 32) {
#pragma unroll
    for (int h = 0; h < 2; ++h) {
      int q = pp + 8 * h;
      unsigned* dg = &lBg[(4 * c4) * 17 + q];
      unsigned* du = &lBu[(4 * c4) * 17 + q];
      dg[0]  = pack2bf(rg[h][0].x, rg[h][1].x); dg[17] = pack2bf(rg[h][0].y, rg[h][1].y);
      dg[34] = pack2bf(rg[h][0].z, rg[h][1].z); dg[51] = pack2bf(rg[h][0].w, rg[h][1].w);
      du[0]  = pack2bf(ru[h][0].x, ru[h][1].x); du[17] = pack2bf(ru[h][0].y, ru[h][1].y);
      du[34] = pack2bf(ru[h][0].z, ru[h][1].z); du[51] = pack2bf(ru[h][0].w, ru[h][1].w);
    }
    __syncthreads();
    if (k0 + 32 < Hh) {
#pragma unroll
      for (int h = 0; h < 2; ++h) {
        const float* gq = g0p + (size_t)(k0 + 32 + 16 * h) * (2 * Ii);
        const float* uq = u0p + (size_t)(k0 + 32 + 16 * h) * (2 * Ii);
        rg[h][0] = *(const float4*)gq; rg[h][1] = *(const float4*)(gq + 2 * Ii);
        ru[h][0] = *(const float4*)uq; ru[h][1] = *(const float4*)(uq + 2 * Ii);
      }
    }
    unsigned au[8], bg[8], bu[8];
    load_afrag_pan(pA, 514, lane, k0 >> 1, au);
    load_bfrag(lBg, ni, lane, bg);
    load_bfrag(lBu, ni, lane, bu);
    cg = wmma_bf16(au, bg, cg);
    cu = wmma_bf16(au, bu, cu);
    __syncthreads();
  }
  int grp = lane >> 4, nn = lane & 15;
  int col = cbase + ni * 16 + nn;
#pragma unroll
  for (int j = 0; j < 8; ++j) {
    int r = j + grp * 8;
    hact[((size_t)e * CAP + mt * 16 + r) * Ii + col] = f2bf(silu(cg[j]) * cu[j]);
  }
}

// ---------------- 6) Routed experts down + weighted scatter-add ----------------
__global__ void __launch_bounds__(256) k_expert_down(const unsigned short* __restrict__ hact,
    const float* __restrict__ w2, const int* __restrict__ counts,
    const int* __restrict__ etok, const float* __restrict__ ewt, float* __restrict__ out) {
  const int NT = Hh / 128;                       // 8
  const int MT = CAP / 16;                       // 64
  int b = blockIdx.x;
  int e = b / (MT * NT); int rem = b % (MT * NT);
  int mt = rem / NT, nt = rem % NT;
  int n = counts[e]; if (n > CAP) n = CAP;
  if (mt * 16 >= n) return;
  int nrem = n - mt * 16; if (nrem > 16) nrem = 16;

  __shared__ unsigned pA[16 * 257];
  __shared__ unsigned lB[128 * 17];
  __shared__ int   sTok[16];
  __shared__ float sWt[16];
  int tid = threadIdx.x, wv = tid >> 5, lane = tid & 31;
  int ni = wv, cbase = nt * 128;
  if (tid < 16) {
    sTok[tid] = (tid < nrem) ? etok[e * CAP + mt * 16 + tid] : -1;
    sWt[tid]  = (tid < nrem) ? ewt[e * CAP + mt * 16 + tid] : 0.f;
  }
  if (wv == 0)
    tdm_load_2d(pA, hact + ((size_t)e * CAP + mt * 16) * Ii, Ii, 16, Ii, lane);
  int c4 = tid & 31, pp = tid >> 5;
  const float* b0p = w2 + (size_t)e * Ii * Hh + (size_t)(2 * pp) * Hh + cbase + 4 * c4;
  float4 rb[2][2];
#pragma unroll
  for (int h = 0; h < 2; ++h) {
    const float* bq = b0p + (size_t)(16 * h) * Hh;
    rb[h][0] = *(const float4*)bq; rb[h][1] = *(const float4*)(bq + Hh);
  }
  if (wv == 0) tdm_wait_all();
  __syncthreads();
  v8f acc = {};
  for (int k0 = 0; k0 < Ii; k0 += 32) {
#pragma unroll
    for (int h = 0; h < 2; ++h) {
      int q = pp + 8 * h;
      unsigned* db = &lB[(4 * c4) * 17 + q];
      db[0]  = pack2bf(rb[h][0].x, rb[h][1].x); db[17] = pack2bf(rb[h][0].y, rb[h][1].y);
      db[34] = pack2bf(rb[h][0].z, rb[h][1].z); db[51] = pack2bf(rb[h][0].w, rb[h][1].w);
    }
    __syncthreads();
    if (k0 + 32 < Ii) {
#pragma unroll
      for (int h = 0; h < 2; ++h) {
        const float* bq = b0p + (size_t)(k0 + 32 + 16 * h) * Hh;
        rb[h][0] = *(const float4*)bq; rb[h][1] = *(const float4*)(bq + Hh);
      }
    }
    unsigned au[8], bu[8];
    load_afrag_pan(pA, 257, lane, k0 >> 1, au);
    load_bfrag(lB, ni, lane, bu);
    acc = wmma_bf16(au, bu, acc);
    __syncthreads();
  }
  int grp = lane >> 4, nn = lane & 15;
  int col = cbase + ni * 16 + nn;
#pragma unroll
  for (int j = 0; j < 8; ++j) {
    int r = j + grp * 8;
    if (r < nrem) {
      atomicAdd(&out[(size_t)sTok[r] * Hh + col], sWt[r] * acc[j]);
    }
  }
}

// ---------------- launcher ----------------
extern "C" void kernel_launch(void* const* d_in, const int* in_sizes, int n_in,
                              void* d_out, int out_size, void* d_ws, size_t ws_size,
                              hipStream_t stream) {
  const float* hs     = (const float*)d_in[0];
  const float* rms_w  = (const float*)d_in[1];
  const float* gate_w = (const float*)d_in[2];
  const float* w13    = (const float*)d_in[3];
  const float* w2     = (const float*)d_in[4];
  const float* sw_g   = (const float*)d_in[5];
  const float* sw_u   = (const float*)d_in[6];
  const float* sw_d   = (const float*)d_in[7];
  float* out = (float*)d_out;

  const int T = in_sizes[0] / Hh;                // 8192 tokens

  char* p = (char*)d_ws;
  unsigned* xb16 = (unsigned*)p;             p += (size_t)T * Hh * 2;
  unsigned short* hact = (unsigned short*)p; p += (size_t)Ee * CAP * Ii * 2;
  unsigned short* hsh  = (unsigned short*)p; p += (size_t)T * Ii * 2;
  int* counts = (int*)p;                     p += 256;
  int* etok   = (int*)p;                     p += (size_t)Ee * CAP * 4;
  float* ewt  = (float*)p;                   p += (size_t)Ee * CAP * 4;

  k_rmsnorm<<<T, 256, 0, stream>>>(hs, rms_w, xb16);
  k_zero_counts<<<1, 64, 0, stream>>>(counts);
  k_gate<<<T / 8, 256, 0, stream>>>(xb16, gate_w, counts, etok, ewt, T);
  k_shared_up<<<(T / 16) * (Ii / 128), 256, 0, stream>>>(xb16, sw_g, sw_u, hsh);
  k_shared_down<<<(T / 16) * (Hh / 128), 256, 0, stream>>>(hsh, sw_d, hs, out);
  k_expert_up<<<Ee * (CAP / 16) * (Ii / 128), 256, 0, stream>>>(xb16, w13, counts, etok, hact, T);
  k_expert_down<<<Ee * (CAP / 16) * (Hh / 128), 256, 0, stream>>>(hact, w2, counts, etok, ewt, out);
}
